// KeyValueMemory_78967268704405
// MI455X (gfx1250) — matrix operand order repaired and checked by reading, not compiled, not run on hardware
//
#include <hip/hip_runtime.h>
#include <hip/hip_bf16.h>
#include <stdint.h>

// ---------------------------------------------------------------------------
// Problem constants (reference: B=8192, M=32, K=V=512)
// ---------------------------------------------------------------------------
#define BB   8192
#define MM   32
#define KK   512
#define VV   512
#define TB   16          // batches per workgroup (one WMMA M-tile)
#define NTHREADS 256     // 8 waves of 32

typedef __attribute__((ext_vector_type(16))) __bf16 v16bf;
typedef __attribute__((ext_vector_type(8)))  float  v8f;
typedef __attribute__((ext_vector_type(4)))  float  f4v;   // clang vector: OK for nontemporal builtins

union V16 { v16bf v; uint4 q[2]; };

// fp32 -> bf16 round-to-nearest-even (raw bits)
__device__ __forceinline__ uint16_t f2bf(float f) {
    uint32_t u = __float_as_uint(f);
    u = (u + 0x7FFFu + ((u >> 16) & 1u)) >> 16;
    return (uint16_t)u;
}

// Non-temporal float4 load for read-once streams (lowers to TH=NT on gfx1250).
__device__ __forceinline__ f4v ldnt4(const float* p) {
    return __builtin_nontemporal_load((const f4v*)p);
}

// ---------------------------------------------------------------------------
// Weight conversion: Wq^T, Wk (as-is), Wv^T, Wo^T  ->  bf16 in workspace.
// The WMMA B-fragment wants per-lane a COLUMN of W contiguously, i.e. a row
// of W^T.  For the qk = Wk @ q stage the effective B is Wk^T, whose transpose
// is Wk itself, so Wk is stored untransposed.
// ---------------------------------------------------------------------------
__global__ __launch_bounds__(256) void convert_weights_kernel(
    const float* __restrict__ Wq, const float* __restrict__ Wk,
    const float* __restrict__ Wv, const float* __restrict__ Wo,
    uint16_t* __restrict__ ws)
{
    int idx = blockIdx.x * 256 + threadIdx.x;       // 0 .. 512*512-1
    int i = idx >> 9;                               // input row  (in-dim)
    int j = idx & 511;                              // input col  (out-dim)
    uint16_t* WqT = ws;
    uint16_t* WkB = ws + 262144;
    uint16_t* WvT = ws + 524288;
    uint16_t* WoT = ws + 786432;
    WqT[j * 512 + i] = f2bf(Wq[idx]);
    WkB[idx]         = f2bf(Wk[idx]);
    WvT[j * 512 + i] = f2bf(Wv[idx]);
    WoT[j * 512 + i] = f2bf(Wo[idx]);
}

// ---------------------------------------------------------------------------
// One [16 x 512] = [16 x 512] * [512 x 512] GEMM, bf16 WMMA, f32 accum.
// A: LDS, row-major bf16 [16][512].  Bt: global bf16, row n = column n of W.
// Each wave owns 4 N-tiles of 16 columns (8 waves * 4 * 16 = 512).
// A-fragment (16x32, 16-bit): lane l+16h -> row l; elems e -> K = k0 + 16h + e
// (contiguous, same as B).  B-fragment (32x16): lane l+16h -> col l,
// elems e -> K = k0 + 16h + e (one contiguous 32B run).
// ---------------------------------------------------------------------------
__device__ __forceinline__ void gemm_16x512(
    const uint16_t* __restrict__ Bt,   // global bf16 [512][512] (L2-resident)
    const uint16_t* Abuf,              // LDS bf16 [16][512]
    v8f acc[4], int wave, int l, int h)
{
    for (int k0 = 0; k0 < KK; k0 += 32) {
        V16 a;
        a.q[0] = *(const uint4*)(Abuf + l * KK + k0 + 16 * h);
        a.q[1] = *(const uint4*)(Abuf + l * KK + k0 + 16 * h + 8);
#pragma unroll
        for (int t = 0; t < 4; ++t) {
            int n0 = (wave * 4 + t) * 16;
            const uint4* pb = (const uint4*)(Bt + (size_t)(n0 + l) * KK + k0 + 16 * h);
            V16 b;
            b.q[0] = pb[0];
            b.q[1] = pb[1];
            acc[t] = __builtin_amdgcn_wmma_f32_16x16x32_bf16(
                false, a.v, false, b.v, (short)0, acc[t], false, false);
        }
    }
}

// ---------------------------------------------------------------------------
// Fused kernel: one workgroup handles 16 batches end-to-end.
// ---------------------------------------------------------------------------
__global__ __launch_bounds__(NTHREADS) void kvmem_fused_kernel(
    const float* __restrict__ mem_keys,    // [B,M,K]
    const float* __restrict__ mem_values,  // [B,M,V]
    const unsigned char* __restrict__ mem_mask,   // [B,M] bool
    const int* __restrict__ position,      // [B]
    const float* __restrict__ new_keys,    // [B,K]
    const float* __restrict__ new_values,  // [B,V]
    const unsigned char* __restrict__ write_mask, // [B] bool
    const float* __restrict__ query,       // [B,K]
    const uint16_t* __restrict__ WqT,      // bf16 [512][512]
    const uint16_t* __restrict__ WkB,
    const uint16_t* __restrict__ WvT,
    const uint16_t* __restrict__ WoT,
    const float* __restrict__ bq,
    const float* __restrict__ bv,
    const float* __restrict__ bo,
    const float* __restrict__ gq, const float* __restrict__ bq_ln,
    const float* __restrict__ go, const float* __restrict__ bo_ln,
    float* __restrict__ out)               // [B,V]
{
    __shared__ uint16_t Abuf[TB * KK];     // bf16 activations (A matrix)   16KB
    __shared__ float    Fbuf[TB * KK];     // f32 staging (qk / pre-LN out) 32KB
    __shared__ float    sc[TB * MM];       // scores -> softmax weights      2KB

    const int tid  = threadIdx.x;
    const int wave = tid >> 5;
    const int lane = tid & 31;
    const int l    = lane & 15;
    const int h    = lane >> 4;
    const int B0   = blockIdx.x * TB;

    // ---- S1: LayerNorm(query) -> Abuf (bf16). 16 lanes per row. ----
    {
        int row = tid >> 4;
        int c0  = tid & 15;
        const float* qrow = query + (size_t)(B0 + row) * KK;
        float x[32];
        float s = 0.f, ss = 0.f;
#pragma unroll
        for (int j = 0; j < 8; ++j) {                 // 8 NT float4 loads
            f4v v = ldnt4(qrow + (c0 + 16 * j) * 4 - c0 * 4 + c0 * 4); // placeholder pattern replaced below
            (void)v;
        }
        // strided scalar pattern (keeps 16-lane row ownership); NT via vector
        // loads is awkward with stride-16, so plain loads here (tiny: 32 KB/WG)
        s = 0.f; ss = 0.f;
#pragma unroll
        for (int j = 0; j < 32; ++j) {
            float v = qrow[c0 + 16 * j];
            x[j] = v; s += v; ss += v * v;
        }
#pragma unroll
        for (int m = 8; m >= 1; m >>= 1) {
            s  += __shfl_xor(s,  m, 16);
            ss += __shfl_xor(ss, m, 16);
        }
        float mu   = s * (1.f / KK);
        float var  = ss * (1.f / KK) - mu * mu;
        float rstd = rsqrtf(var + 1e-5f);
#pragma unroll
        for (int j = 0; j < 32; ++j) {
            int c = c0 + 16 * j;
            Abuf[row * KK + c] = f2bf((x[j] - mu) * rstd * gq[c] + bq_ln[c]);
        }
    }
    __syncthreads();

    // ---- S2a: q = qn @ Wq + bq  ->  Abuf (bf16) ----
    {
        v8f acc[4] = {};
        gemm_16x512(WqT, Abuf, acc, wave, l, h);
        __syncthreads();                       // all reads of Abuf done
#pragma unroll
        for (int t = 0; t < 4; ++t) {
            int n0 = (wave * 4 + t) * 16;
            float bias = bq[n0 + l];
#pragma unroll
            for (int r = 0; r < 8; ++r)
                Abuf[(r + 8 * h) * KK + n0 + l] = f2bf(acc[t][r] + bias);
        }
    }
    __syncthreads();

    // ---- S2b: qk = q @ Wk^T  ->  Fbuf (f32).  (bk cancels in softmax) ----
    {
        v8f acc[4] = {};
        gemm_16x512(WkB, Abuf, acc, wave, l, h);
#pragma unroll
        for (int t = 0; t < 4; ++t) {
            int n0 = (wave * 4 + t) * 16;
#pragma unroll
            for (int r = 0; r < 8; ++r)
                Fbuf[(r + 8 * h) * KK + n0 + l] = acc[t][r];
        }
    }
    __syncthreads();

    // ---- S3: scores[b][m] = <keys_scattered, qk> / sqrt(K), masked ----
    // keys are read exactly once across the whole dispatch -> non-temporal.
    for (int p = tid; p < TB * MM; p += NTHREADS) {
        int b = p >> 5, m = p & 31;
        int gb = B0 + b;
        int pos = position[gb];
        bool wm = write_mask[gb] != 0;
        bool slot = wm && (pos == m);
        const float* krow = slot ? (new_keys + (size_t)gb * KK)
                                 : (mem_keys + ((size_t)gb * MM + m) * KK);
        const float* q4 = Fbuf + b * KK;
        float s = 0.f;
#pragma unroll 8
        for (int j = 0; j < KK / 4; ++j) {
            f4v a = ldnt4(krow + 4 * j);
            const f4v c = *(const f4v*)(q4 + 4 * j);
            s += a.x * c.x + a.y * c.y + a.z * c.z + a.w * c.w;
        }
        bool valid = (mem_mask[(size_t)gb * MM + m] != 0) || slot;
        sc[b * MM + m] = valid ? s * 0.04419417382415922f : -1e9f;  // 1/sqrt(512)
    }
    __syncthreads();

    // ---- S4: softmax over M=32 per batch (wave-wide shuffles, lane = m) ----
    for (int bb = 0; bb < 2; ++bb) {
        int b = wave * 2 + bb;
        float v = sc[b * MM + lane];
        float mx = v;
#pragma unroll
        for (int m = 16; m >= 1; m >>= 1) mx = fmaxf(mx, __shfl_xor(mx, m));
        float e = __expf(v - mx);
        float sum = e;
#pragma unroll
        for (int m = 16; m >= 1; m >>= 1) sum += __shfl_xor(sum, m);
        sc[b * MM + lane] = e * __frcp_rn(sum);
    }
    __syncthreads();

    // ---- S5: r = sum_m w[m] * values_scattered[m]  ->  Abuf (bf16) ----
    // values also read exactly once -> non-temporal.
    {
        int b = tid >> 4, c16 = tid & 15;
        int gb = B0 + b;
        int pos = position[gb];
        bool wm = write_mask[gb] != 0;
        f4v a[8] = {};
        for (int m = 0; m < MM; ++m) {
            float wgt = sc[b * MM + m];
            bool slot = wm && (pos == m);
            const float* vrow = slot ? (new_values + (size_t)gb * VV)
                                     : (mem_values + ((size_t)gb * MM + m) * VV);
#pragma unroll
            for (int jj = 0; jj < 8; ++jj) {
                f4v t4 = ldnt4(vrow + (c16 + 16 * jj) * 4);
                a[jj] += wgt * t4;
            }
        }
#pragma unroll
        for (int jj = 0; jj < 8; ++jj) {
            int c = (c16 + 16 * jj) * 4;
            Abuf[b * KK + c + 0] = f2bf(a[jj].x);
            Abuf[b * KK + c + 1] = f2bf(a[jj].y);
            Abuf[b * KK + c + 2] = f2bf(a[jj].z);
            Abuf[b * KK + c + 3] = f2bf(a[jj].w);
        }
    }
    __syncthreads();

    // ---- S6: retrieved = r @ Wv + bv  ->  Abuf (bf16) ----
    {
        v8f acc[4] = {};
        gemm_16x512(WvT, Abuf, acc, wave, l, h);
        __syncthreads();
#pragma unroll
        for (int t = 0; t < 4; ++t) {
            int n0 = (wave * 4 + t) * 16;
            float bias = bv[n0 + l];
#pragma unroll
            for (int r = 0; r < 8; ++r)
                Abuf[(r + 8 * h) * KK + n0 + l] = f2bf(acc[t][r] + bias);
        }
    }
    __syncthreads();

    // ---- S7: pre = retrieved @ Wo + bo  ->  Fbuf (f32) ----
    {
        v8f acc[4] = {};
        gemm_16x512(WoT, Abuf, acc, wave, l, h);
#pragma unroll
        for (int t = 0; t < 4; ++t) {
            int n0 = (wave * 4 + t) * 16;
            float bias = bo[n0 + l];
#pragma unroll
            for (int r = 0; r < 8; ++r)
                Fbuf[(r + 8 * h) * KK + n0 + l] = acc[t][r] + bias;
        }
    }
    __syncthreads();

    // ---- S8: final LayerNorm -> global out (write-once -> NT stores) ----
    {
        int row = tid >> 4;
        int c0  = tid & 15;
        float x[32];
        float s = 0.f, ss = 0.f;
#pragma unroll
        for (int j = 0; j < 32; ++j) {
            float v = Fbuf[row * KK + c0 + 16 * j];
            x[j] = v; s += v; ss += v * v;
        }
#pragma unroll
        for (int m = 8; m >= 1; m >>= 1) {
            s  += __shfl_xor(s,  m, 16);
            ss += __shfl_xor(ss, m, 16);
        }
        float mu   = s * (1.f / VV);
        float var  = ss * (1.f / VV) - mu * mu;
        float rstd = rsqrtf(var + 1e-5f);
        float* orow = out + (size_t)(B0 + row) * VV;
#pragma unroll
        for (int j = 0; j < 32; ++j) {
            int c = c0 + 16 * j;
            __builtin_nontemporal_store((x[j] - mu) * rstd * go[c] + bo_ln[c],
                                        orow + c);
        }
    }
}

// ---------------------------------------------------------------------------
extern "C" void kernel_launch(void* const* d_in, const int* in_sizes, int n_in,
                              void* d_out, int out_size, void* d_ws, size_t ws_size,
                              hipStream_t stream) {
    const float* mem_keys     = (const float*)d_in[0];
    const float* mem_values   = (const float*)d_in[1];
    const unsigned char* mem_mask   = (const unsigned char*)d_in[2];
    const int*   position     = (const int*)d_in[3];
    const float* new_keys     = (const float*)d_in[4];
    const float* new_values   = (const float*)d_in[5];
    const unsigned char* write_mask = (const unsigned char*)d_in[6];
    const float* query        = (const float*)d_in[7];
    const float* Wq = (const float*)d_in[8];
    const float* bq = (const float*)d_in[9];
    const float* Wk = (const float*)d_in[10];
    /* bk (d_in[11]) cancels in the softmax and is unused */
    const float* Wv = (const float*)d_in[12];
    const float* bv = (const float*)d_in[13];
    const float* Wo = (const float*)d_in[14];
    const float* bo = (const float*)d_in[15];
    const float* gq    = (const float*)d_in[16];
    const float* bq_ln = (const float*)d_in[17];
    const float* go    = (const float*)d_in[18];
    const float* bo_ln = (const float*)d_in[19];
    float* out = (float*)d_out;

    uint16_t* ws = (uint16_t*)d_ws;            // 4 x 512KB bf16 weights
    uint16_t* WqT = ws;
    uint16_t* WkB = ws + 262144;
    uint16_t* WvT = ws + 524288;
    uint16_t* WoT = ws + 786432;

    convert_weights_kernel<<<(512 * 512) / 256, 256, 0, stream>>>(Wq, Wk, Wv, Wo, ws);

    kvmem_fused_kernel<<<BB / TB, NTHREADS, 0, stream>>>(
        mem_keys, mem_values, mem_mask, position, new_keys, new_values,
        write_mask, query, WqT, WkB, WvT, WoT,
        bq, bv, bo, gq, bq_ln, go, bo_ln, out);
}
